// FChCombxValEncoder_88802743812298
// MI455X (gfx1250) — compile-verified, matrix operand order checked by reading, not run on hardware
//
#include <hip/hip_runtime.h>
#include <hip/hip_bf16.h>
#include <stdint.h>
#include <string.h>

// ---------------- problem constants (from reference) ----------------
#define MAXV      52000.0f
#define MINV     (-53000.0f)
#define NLEVELS   1000
#define D_OUT     10000      // hypervector dimension
#define NF        4096       // TIMESTAMPS * CHANNELS features
#define LNG       (NF - 3)   // number of 4-grams = 4093 (odd -> sum never 0)
#define PITCH     10016      // bytes per packed row: [4B wrap prefix][10000 cols][12 pad]
#define NTILES    (D_OUT / 16)   // 625 WMMA column tiles

typedef __attribute__((ext_vector_type(8))) int v8i;

__device__ __forceinline__ uint32_t ldu32(const uint8_t* p) {
    uint32_t v;
    __builtin_memcpy(&v, p, 4);   // unaligned-safe 32-bit load
    return v;
}

// --------------------------------------------------------------------
// Kernel 1: level-index gather + MAP bind + sign pack (HBM-bound pass).
// For feature row i: idx = quantize(x[i]); byte(c) = signbit(signals[idx][c])
// XOR signbit(feat[i][c])  (0 => +1, 1 => -1).  Row layout adds a 4-byte
// prefix holding columns 9996..9999 so the mod-10000 column rotation in the
// n-gram kernel becomes a plain negative address offset.
// Reads ~328 MB of floats once (40 MB distinct signal rows stay in L2),
// writes 41 MB of packed bytes -> this pass sets the ~9us roofline, so use
// b128 loads + packed dword stores for maximum per-instruction bytes.
// --------------------------------------------------------------------
__global__ __launch_bounds__(256)
void bind_pack_kernel(const float* __restrict__ x,
                      const float* __restrict__ signals,
                      const float* __restrict__ feat,
                      uint8_t* __restrict__ P) {
    const int i = blockIdx.x;                 // feature row 0..4095
    // Level.value_to_index: round-half-even then clip (matches jnp.round)
    float t = (x[i] - MINV) / (MAXV - MINV) * (float)(NLEVELS - 1);
    int idx = (int)rintf(t);
    idx = idx < 0 ? 0 : (idx > NLEVELS - 1 ? NLEVELS - 1 : idx);

    const float4* srow = (const float4*)(signals + (size_t)idx * D_OUT);  // 16B-aligned
    const float4* frow = (const float4*)(feat    + (size_t)i   * D_OUT);  // 16B-aligned
    uint8_t*      prow = P + (size_t)i * PITCH;
    uint32_t*   prow32 = (uint32_t*)(prow + 4);                           // 4B-aligned

    for (int g = threadIdx.x; g < D_OUT / 4; g += 256) {   // 2500 groups of 4 cols
        float4 s = srow[g];
        float4 f = frow[g];
        // sign(s)*sign(f): byte = XOR of IEEE sign bits (entries are exactly +/-1.0)
        uint32_t b0 = (__float_as_uint(s.x) ^ __float_as_uint(f.x)) >> 31;
        uint32_t b1 = (__float_as_uint(s.y) ^ __float_as_uint(f.y)) >> 31;
        uint32_t b2 = (__float_as_uint(s.z) ^ __float_as_uint(f.z)) >> 31;
        uint32_t b3 = (__float_as_uint(s.w) ^ __float_as_uint(f.w)) >> 31;
        uint32_t packed = b0 | (b1 << 8) | (b2 << 16) | (b3 << 24);
        prow32[g] = packed;
        if (g == D_OUT / 4 - 1) {            // wrap prefix = cols 9996..9999
            *(uint32_t*)prow = packed;
        }
    }
}

// --------------------------------------------------------------------
// Kernel 2: n-gram parity + bundle via V_WMMA_I32_16X16X64_IU8 (L2-resident).
// ngbit(i,d) = P[i][d-3] ^ P[i+1][d-2] ^ P[i+2][d-1] ^ P[i+3][d]  (parity of
// the 4-gram product).  Bundle sum = LNG - 2 * count, count = sum_i ngbit.
// count is a ones(16x64) x B(64x16, u8 0/1) matmul accumulated over 64 row
// chunks -> one integer WMMA per chunk per 16-column tile.
// Per lane, w(j) = 4-byte window [d-3..d] of row j; ng(i) = byte0 of
//   w(i) ^ (w(i+1)>>8) ^ (w(i+2)>>16) ^ (w(i+3)>>24).
// B layout (64x16 u8, 8 VGPRs, wave32): lanes 0-15 hold K0-15 (v0-3) and
// K32-47 (v4-7); lanes 16-31 hold K16-31 and K48-63; bytes = consecutive K.
// --------------------------------------------------------------------
__global__ __launch_bounds__(256)
void ngram_bundle_wmma_kernel(const uint8_t* __restrict__ P,
                              float* __restrict__ out) {
    const int lane = threadIdx.x & 31;
    const int tile = blockIdx.x * 8 + (threadIdx.x >> 5);   // one wave per 16-col tile
    if (tile >= NTILES) return;                             // wave-uniform exit

    const int n    = lane & 15;        // column within tile (N index)
    const int half = lane >> 4;        // 0: K 0-15 & 32-47 ; 1: K 16-31 & 48-63
    const int col  = tile * 16 + n;    // absolute output column d
    const uint8_t* base = P + (4 + col - 3);   // window start: column d-3 (prefix covers d<3)

    v8i acc = {0, 0, 0, 0, 0, 0, 0, 0};
    v8i onesA;                                   // A = 16x64 matrix of u8 ones
#pragma unroll
    for (int v = 0; v < 8; ++v) onesA[v] = 0x01010101;

    for (int chunk = 0; chunk < NF; chunk += 64) {
        const int rlo = chunk + half * 16;       // rows feeding this lane's low K block
        const int rhi = rlo + 32;                // rows feeding the high K block
        uint32_t wlo[19], whi[19];               // 16 K values need rows +0..+18 (window +3)
#pragma unroll
        for (int t = 0; t < 19; ++t) {
            int jl = rlo + t; jl = jl > NF - 1 ? NF - 1 : jl;   // clamp tail (feeds only i>=LNG)
            int jh = rhi + t; jh = jh > NF - 1 ? NF - 1 : jh;
            wlo[t] = ldu32(base + (size_t)jl * PITCH);
            whi[t] = ldu32(base + (size_t)jh * PITCH);
        }
        v8i B;
#pragma unroll
        for (int v = 0; v < 4; ++v) {
            uint32_t rL = 0, rH = 0;
#pragma unroll
            for (int b = 0; b < 4; ++b) {
                const int t  = v * 4 + b;        // K offset within the 16-row block
                uint32_t zL = wlo[t] ^ (wlo[t + 1] >> 8) ^ (wlo[t + 2] >> 16) ^ (wlo[t + 3] >> 24);
                uint32_t zH = whi[t] ^ (whi[t + 1] >> 8) ^ (whi[t + 2] >> 16) ^ (whi[t + 3] >> 24);
                uint32_t ngL = (rlo + t < LNG) ? (zL & 0xFFu) : 0u;   // select, no divergence
                uint32_t ngH = (rhi + t < LNG) ? (zH & 0xFFu) : 0u;
                rL |= ngL << (8 * b);
                rH |= ngH << (8 * b);
            }
            B[v]     = (int)rL;
            B[4 + v] = (int)rH;
        }
        // D[m][n] = sum_k 1 * B[k][n]  -> every row m holds count for column n
        acc = __builtin_amdgcn_wmma_i32_16x16x64_iu8(false, onesA, false, B, acc,
                                                     false, false);
    }

    const int count = acc[0];                     // identical across M rows
    if (lane < 16) {                              // lanes 16-31 hold duplicate columns
        out[col] = ((LNG - 2 * count) > 0) ? 1.0f : -1.0f;   // hard_quantize
    }
}

// --------------------------------------------------------------------
extern "C" void kernel_launch(void* const* d_in, const int* in_sizes, int n_in,
                              void* d_out, int out_size, void* d_ws, size_t ws_size,
                              hipStream_t stream) {
    (void)in_sizes; (void)n_in; (void)out_size; (void)ws_size;
    const float* x       = (const float*)d_in[0];   // (128,32) flattened = 4096
    const float* signals = (const float*)d_in[1];   // (1000, 10000)
    const float* feat    = (const float*)d_in[2];   // (4096, 10000)
    float*   out = (float*)d_out;                   // (1, 10000) float32
    uint8_t* P   = (uint8_t*)d_ws;                  // needs NF*PITCH = 41,025,536 B

    bind_pack_kernel<<<NF, 256, 0, stream>>>(x, signals, feat, P);
    ngram_bundle_wmma_kernel<<<(NTILES + 7) / 8, 256, 0, stream>>>(P, out);
}
